// BrainNetGIN_88742614270023
// MI455X (gfx1250) — matrix operand order repaired and checked by reading, not compiled, not run on hardware
//
#include <hip/hip_runtime.h>

typedef __attribute__((ext_vector_type(16))) _Float16 v16h;
typedef __attribute__((ext_vector_type(8)))  _Float16 v8h;
typedef __attribute__((ext_vector_type(8)))  float    v8f;

#define N_NODES   118784
#define N_EDGES   3000000
#define IN_DIM    116
#define EMB_DIM   16
#define KPAD      160      // 132 padded to 5 chunks of 32
#define HID       64
#define ED_DIM    5
#define NGRAPH    1024
#define ROWS_PER_G 116
#define NLAYER    3

// ---------------------------------------------------------------------------
// WMMA helpers (CDNA5: V_WMMA_F32_16X16X32_F16, D = A(16x32) * B(32x16) + C)
// A frag: lane L -> row M = L%16; half = L/16; a[0..7]=K[half*8 .. +7],
//         a[8..15]=K[16+half*8 .. +7]
// B frag: lane L -> col N = n0+L%16; b[0..7]=K[half*8 .. +7], b[8..15]=K[16+half*8 .. +7]
// C/D    : lane L -> col N = L%16; c[r] = C[M = r + 8*(L/16)][N]
// ---------------------------------------------------------------------------
__device__ __forceinline__ v8f wmma_f16(v16h a, v16h b, v8f c) {
  return __builtin_amdgcn_wmma_f32_16x16x32_f16(
      false, a, false, b, (short)0, c, false, false);
}

// Packed f16 weights, zero-padded: column n0+lane%16, K in A's table order, no guards.
__device__ __forceinline__ v16h load_b_packed(const _Float16* __restrict__ Wp,
                                              int k0, int n0, int lane) {
  v16h b;
  const int n = n0 + (lane & 15);
  const int kb = k0 + (lane >> 4) * 8;
#pragma unroll
  for (int s = 0; s < 8; ++s) {
    b[s] = Wp[(size_t)(kb + s) * HID + n];
    b[8 + s] = Wp[(size_t)(kb + 16 + s) * HID + n];
  }
  return b;
}

__device__ __forceinline__ v16h make_a(v8h lo, v8h hi) {
  return __builtin_shufflevector(lo, hi, 0, 1, 2, 3, 4, 5, 6, 7,
                                 8, 9, 10, 11, 12, 13, 14, 15);
}

// ---------------------------------------------------------------------------
// Pack kernels: move all padding/conversion out of the hot GEMMs.
// ---------------------------------------------------------------------------
// xp[N][160] f16 = concat(x, emb[gid]) zero-padded. One block per row.
__global__ void pack_x_kernel(const float* __restrict__ x,
                              const int* __restrict__ gid,
                              const float* __restrict__ emb,
                              _Float16* __restrict__ xp) {
  const int row = blockIdx.x;
  const int c = threadIdx.x;  // 160 threads
  const int g = gid[row];
  float v = (c < IN_DIM) ? x[(size_t)row * IN_DIM + c]
          : (c < IN_DIM + EMB_DIM) ? emb[g * EMB_DIM + (c - IN_DIM)]
          : 0.0f;
  xp[(size_t)row * KPAD + c] = (_Float16)v;
}

// Wp[kpad][64] f16, zero rows past krows. Grid = kpad blocks of 64.
__global__ void pack_w_kernel(const float* __restrict__ W,
                              _Float16* __restrict__ Wp, int krows) {
  const int k = blockIdx.x;
  const int c = threadIdx.x;
  float v = (k < krows) ? W[(size_t)k * HID + c] : 0.0f;
  Wp[(size_t)k * HID + c] = (_Float16)v;
}

// ---------------------------------------------------------------------------
// Encoder: h = relu( xp @ Wenc_p + b_enc )   [N,160]x[160,64] -> [N,64]
// One wave per 16-row tile; K fully unrolled (5 chunks of 32): 20 WMMAs.
// ---------------------------------------------------------------------------
__global__ void encoder_kernel(const _Float16* __restrict__ xp,
                               const _Float16* __restrict__ Wp,
                               const float* __restrict__ benc,
                               float* __restrict__ hout) {
  const int lane = threadIdx.x & 31;
  const int wave = threadIdx.x >> 5;
  const int tile = blockIdx.x * (blockDim.x >> 5) + wave;
  const int row0 = tile * 16;
  const int mrow = row0 + (lane & 15);
  const int base = (lane >> 4) * 8;
  const int half = lane >> 4;
  const int nlo  = lane & 15;

  const _Float16* xr = xp + (size_t)mrow * KPAD;

  v8f acc[4] = {};
#pragma unroll
  for (int k0 = 0; k0 < KPAD; k0 += 32) {
    v8h lo = *(const v8h*)(xr + k0 + base);
    v8h hi = *(const v8h*)(xr + k0 + 16 + base);
    v16h a = make_a(lo, hi);
#pragma unroll
    for (int nt = 0; nt < 4; ++nt)
      acc[nt] = wmma_f16(a, load_b_packed(Wp, k0, nt * 16, lane), acc[nt]);
  }

#pragma unroll
  for (int nt = 0; nt < 4; ++nt) {
#pragma unroll
    for (int r = 0; r < 8; ++r) {
      const int orow = row0 + r + 8 * half;
      const int ocol = nt * 16 + nlo;
      float v = acc[nt][r] + benc[ocol];
      hout[(size_t)orow * HID + ocol] = v > 0.0f ? v : 0.0f;
    }
  }
}

// ---------------------------------------------------------------------------
// Per-edge message: agg[dst] += relu( h[src] + (edge_attr @ We + be) )
// Edge-linear (5x64) fused in registers. One wave per edge, 2 ch/lane.
// ---------------------------------------------------------------------------
__global__ void msg_kernel(const float* __restrict__ h,
                           const int* __restrict__ eidx,
                           const float* __restrict__ eattr,
                           const float* __restrict__ We,
                           const float* __restrict__ be,
                           float* __restrict__ agg) {
  const int lane = threadIdx.x & 31;
  const int e = (blockIdx.x * blockDim.x + threadIdx.x) >> 5;
  if (e >= N_EDGES) return;

  const int src = eidx[e];
  const int dst = eidx[N_EDGES + e];
  const int c0 = lane * 2;

  float ea[ED_DIM];
#pragma unroll
  for (int d = 0; d < ED_DIM; ++d) ea[d] = eattr[(size_t)e * ED_DIM + d];

  float m0 = be[c0], m1 = be[c0 + 1];
#pragma unroll
  for (int d = 0; d < ED_DIM; ++d) {
    m0 += ea[d] * We[d * HID + c0];
    m1 += ea[d] * We[d * HID + c0 + 1];
  }

  const float2 hs = *(const float2*)(h + (size_t)src * HID + c0);
  m0 += hs.x;
  m1 += hs.y;
  m0 = m0 > 0.0f ? m0 : 0.0f;
  m1 = m1 > 0.0f ? m1 : 0.0f;

  atomicAdd(agg + (size_t)dst * HID + c0, m0);
  atomicAdd(agg + (size_t)dst * HID + c0 + 1, m1);
}

// ---------------------------------------------------------------------------
// Fused GINE MLP: out = relu( (h+agg) @ W1 + b1 ) @ W2 + b2   (both 64x64)
// One wave per 16-row tile; hidden tile staged through LDS as f16 (stride 80
// halves => 16B-aligned v8h loads for the second GEMM's A fragments).
// ---------------------------------------------------------------------------
__global__ void gine_mlp_kernel(const float* __restrict__ h,
                                const float* __restrict__ agg,
                                const _Float16* __restrict__ W1p,
                                const float* __restrict__ b1,
                                const _Float16* __restrict__ W2p,
                                const float* __restrict__ b2,
                                float* __restrict__ out, int relu_out) {
  __shared__ _Float16 tls[4][16][80];

  const int lane = threadIdx.x & 31;
  const int wave = threadIdx.x >> 5;
  const int tile = blockIdx.x * (blockDim.x >> 5) + wave;
  const int row0 = tile * 16;
  const int mrow = row0 + (lane & 15);
  const int base = (lane >> 4) * 8;
  const int half = lane >> 4;
  const int nlo  = lane & 15;

  const float* hr = h + (size_t)mrow * HID;
  const float* ar = agg + (size_t)mrow * HID;

  // ---- GEMM1: z = h + agg ; t = relu(z @ W1 + b1)
  v8f acc[4] = {};
#pragma unroll
  for (int k0 = 0; k0 < HID; k0 += 32) {
    float4 p0 = *(const float4*)(hr + k0 + base);
    float4 p1 = *(const float4*)(hr + k0 + base + 4);
    float4 p2 = *(const float4*)(hr + k0 + 16 + base);
    float4 p3 = *(const float4*)(hr + k0 + 16 + base + 4);
    float4 q0 = *(const float4*)(ar + k0 + base);
    float4 q1 = *(const float4*)(ar + k0 + base + 4);
    float4 q2 = *(const float4*)(ar + k0 + 16 + base);
    float4 q3 = *(const float4*)(ar + k0 + 16 + base + 4);
    v16h a;
    a[0]  = (_Float16)(p0.x + q0.x); a[1]  = (_Float16)(p0.y + q0.y);
    a[2]  = (_Float16)(p0.z + q0.z); a[3]  = (_Float16)(p0.w + q0.w);
    a[4]  = (_Float16)(p1.x + q1.x); a[5]  = (_Float16)(p1.y + q1.y);
    a[6]  = (_Float16)(p1.z + q1.z); a[7]  = (_Float16)(p1.w + q1.w);
    a[8]  = (_Float16)(p2.x + q2.x); a[9]  = (_Float16)(p2.y + q2.y);
    a[10] = (_Float16)(p2.z + q2.z); a[11] = (_Float16)(p2.w + q2.w);
    a[12] = (_Float16)(p3.x + q3.x); a[13] = (_Float16)(p3.y + q3.y);
    a[14] = (_Float16)(p3.z + q3.z); a[15] = (_Float16)(p3.w + q3.w);
#pragma unroll
    for (int nt = 0; nt < 4; ++nt)
      acc[nt] = wmma_f16(a, load_b_packed(W1p, k0, nt * 16, lane), acc[nt]);
  }

#pragma unroll
  for (int nt = 0; nt < 4; ++nt) {
#pragma unroll
    for (int r = 0; r < 8; ++r) {
      float v = acc[nt][r] + b1[nt * 16 + nlo];
      tls[wave][r + 8 * half][nt * 16 + nlo] = (_Float16)(v > 0.0f ? v : 0.0f);
    }
  }
  __syncthreads();

  // ---- GEMM2: out = t @ W2 + b2
  v8f acc2[4] = {};
#pragma unroll
  for (int k0 = 0; k0 < HID; k0 += 32) {
    v8h lo = *(const v8h*)(&tls[wave][lane & 15][k0 + base]);
    v8h hi = *(const v8h*)(&tls[wave][lane & 15][k0 + 16 + base]);
    v16h a = make_a(lo, hi);
#pragma unroll
    for (int nt = 0; nt < 4; ++nt)
      acc2[nt] = wmma_f16(a, load_b_packed(W2p, k0, nt * 16, lane), acc2[nt]);
  }

#pragma unroll
  for (int nt = 0; nt < 4; ++nt) {
#pragma unroll
    for (int r = 0; r < 8; ++r) {
      const int orow = row0 + r + 8 * half;
      const int ocol = nt * 16 + nlo;
      float v = acc2[nt][r] + b2[ocol];
      if (relu_out) v = v > 0.0f ? v : 0.0f;
      out[(size_t)orow * HID + ocol] = v;
    }
  }
}

// ---------------------------------------------------------------------------
// BatchNorm: pass 1 (per-channel sum / sumsq), pass 2 (scale-shift [+relu]).
// ---------------------------------------------------------------------------
__global__ void fill_zero_kernel(float* __restrict__ p, int n) {
  int i = blockIdx.x * blockDim.x + threadIdx.x;
  if (i < n) p[i] = 0.0f;
}

__global__ void bn_stats_kernel(const float* __restrict__ h,
                                float* __restrict__ stats) {
  const int c = threadIdx.x & 63;
  const int rl = threadIdx.x >> 6;  // 0..3
  const int row0 = blockIdx.x * 256;
  float s = 0.0f, s2 = 0.0f;
  for (int j = rl; j < 256; j += 4) {
    float v = h[(size_t)(row0 + j) * HID + c];
    s += v;
    s2 += v * v;
  }
  atomicAdd(stats + c, s);
  atomicAdd(stats + HID + c, s2);
}

__global__ void bn_apply_kernel(float* __restrict__ h,
                                const float* __restrict__ stats,
                                const float* __restrict__ g,
                                const float* __restrict__ b, int relu) {
  const int i = blockIdx.x * blockDim.x + threadIdx.x;
  if (i >= N_NODES * HID) return;
  const int c = i & (HID - 1);
  const float mean = stats[c] * (1.0f / N_NODES);
  const float var = stats[HID + c] * (1.0f / N_NODES) - mean * mean;
  const float inv = rsqrtf(var + 1e-5f);
  float v = g[c] * (h[i] - mean) * inv + b[c];
  if (relu) v = v > 0.0f ? v : 0.0f;
  h[i] = v;
}

// ---------------------------------------------------------------------------
// Pool (116 contiguous rows per graph) + head MLP (64->116->2).
// ---------------------------------------------------------------------------
__global__ void pool_kernel(const float* __restrict__ h,
                            float* __restrict__ pooled) {
  const int g = blockIdx.x;
  const int c = threadIdx.x;  // 64 threads
  float s = 0.0f;
  for (int j = 0; j < ROWS_PER_G; ++j)
    s += h[(size_t)(g * ROWS_PER_G + j) * HID + c];
  pooled[g * HID + c] = s;
}

__global__ void head_kernel(const float* __restrict__ pooled,
                            const float* __restrict__ Wl1,
                            const float* __restrict__ bl1,
                            const float* __restrict__ Wl2,
                            const float* __restrict__ bl2,
                            float* __restrict__ out) {
  __shared__ float pr[HID];
  __shared__ float mid[IN_DIM];
  const int g = blockIdx.x;
  const int t = threadIdx.x;  // 128 threads
  if (t < HID) {
    float v = pooled[g * HID + t];
    pr[t] = v > 0.0f ? v : 0.0f;
  }
  __syncthreads();
  if (t < IN_DIM) {
    float s = bl1[t];
    for (int c = 0; c < HID; ++c) s += pr[c] * Wl1[c * IN_DIM + t];
    mid[t] = s;
  }
  __syncthreads();
  if (t < 2) {
    float s = bl2[t];
    for (int i = 0; i < IN_DIM; ++i) s += mid[i] * Wl2[i * 2 + t];
    out[g * 2 + t] = s;
  }
}

// ---------------------------------------------------------------------------
extern "C" void kernel_launch(void* const* d_in, const int* in_sizes, int n_in,
                              void* d_out, int out_size, void* d_ws,
                              size_t ws_size, hipStream_t stream) {
  const float* x         = (const float*)d_in[0];
  const float* edge_attr = (const float*)d_in[1];
  const int*   edge_idx  = (const int*)d_in[2];
  // d_in[3] = batch_ids (unused: nodes are contiguous, 116 per graph)
  const int*   group_ids = (const int*)d_in[4];
  const float* emb       = (const float*)d_in[5];
  const float* W_enc     = (const float*)d_in[6];
  const float* b_enc     = (const float*)d_in[7];
  const float* gamma_e   = (const float*)d_in[8];
  const float* beta_e    = (const float*)d_in[9];
  const float* We0       = (const float*)d_in[10];
  const float* be0       = (const float*)d_in[11];
  const float* W10       = (const float*)d_in[12];
  const float* b10       = (const float*)d_in[13];
  const float* W20       = (const float*)d_in[14];
  const float* b20       = (const float*)d_in[15];
  const float* We        = (const float*)d_in[16];
  const float* be        = (const float*)d_in[17];
  const float* W1        = (const float*)d_in[18];
  const float* b1        = (const float*)d_in[19];
  const float* W2        = (const float*)d_in[20];
  const float* b2        = (const float*)d_in[21];
  const float* gamma     = (const float*)d_in[22];
  const float* beta      = (const float*)d_in[23];
  const float* Wl1       = (const float*)d_in[24];
  const float* bl1       = (const float*)d_in[25];
  const float* Wl2       = (const float*)d_in[26];
  const float* bl2       = (const float*)d_in[27];
  float* out = (float*)d_out;

  // Workspace: hA | hB | agg | stats(128) | pooled | xp(f16) | Wenc_p | W*p
  float* ws = (float*)d_ws;
  const size_t NH = (size_t)N_NODES * HID;
  float* hA     = ws;
  float* hB     = hA + NH;
  float* agg    = hB + NH;
  float* stats  = agg + NH;
  float* pooled = stats + 128;
  _Float16* xp     = (_Float16*)(pooled + (size_t)NGRAPH * HID);
  _Float16* wenc_p = xp + (size_t)N_NODES * KPAD;
  _Float16* wmlp_p = wenc_p + (size_t)KPAD * HID;  // 8 x (64x64)

  const int gemm_blocks = N_NODES / 64;  // 4 waves/block, 16 rows/wave
  const int elems       = (int)NH;
  const int zb          = (elems + 255) / 256;
  const int msg_blocks  = N_EDGES / 8;
  const int stat_blocks = N_NODES / 256;

  // ---- one-shot packing (f32 -> padded f16)
  pack_x_kernel<<<N_NODES, KPAD, 0, stream>>>(x, group_ids, emb, xp);
  pack_w_kernel<<<KPAD, HID, 0, stream>>>(W_enc, wenc_p, IN_DIM + EMB_DIM);
  _Float16* w10p = wmlp_p + 0 * HID * HID;
  _Float16* w20p = wmlp_p + 1 * HID * HID;
  pack_w_kernel<<<HID, HID, 0, stream>>>(W10, w10p, HID);
  pack_w_kernel<<<HID, HID, 0, stream>>>(W20, w20p, HID);
  for (int l = 0; l < NLAYER; ++l) {
    pack_w_kernel<<<HID, HID, 0, stream>>>(W1 + (size_t)l * HID * HID,
                                           wmlp_p + (2 + 2 * l) * HID * HID,
                                           HID);
    pack_w_kernel<<<HID, HID, 0, stream>>>(W2 + (size_t)l * HID * HID,
                                           wmlp_p + (3 + 2 * l) * HID * HID,
                                           HID);
  }

  // ---- encoder + BN (no relu after BN per reference)
  encoder_kernel<<<gemm_blocks, 128, 0, stream>>>(xp, wenc_p, b_enc, hA);
  fill_zero_kernel<<<1, 128, 0, stream>>>(stats, 128);
  bn_stats_kernel<<<stat_blocks, 256, 0, stream>>>(hA, stats);
  bn_apply_kernel<<<zb, 256, 0, stream>>>(hA, stats, gamma_e, beta_e, 0);

  // ---- gine0: relu(mlp(h + sum relu(h[src]+lin_e(e))))
  fill_zero_kernel<<<zb, 256, 0, stream>>>(agg, elems);
  msg_kernel<<<msg_blocks, 256, 0, stream>>>(hA, edge_idx, edge_attr, We0, be0,
                                             agg);
  gine_mlp_kernel<<<gemm_blocks, 128, 0, stream>>>(hA, agg, w10p, b10, w20p,
                                                   b20, hB, 1);

  // ---- GINE stack: gine -> BN -> relu, x3
  float* cur = hB;
  float* nxt = hA;
  for (int l = 0; l < NLAYER; ++l) {
    fill_zero_kernel<<<zb, 256, 0, stream>>>(agg, elems);
    msg_kernel<<<msg_blocks, 256, 0, stream>>>(cur, edge_idx, edge_attr,
                                               We + (size_t)l * ED_DIM * HID,
                                               be + (size_t)l * HID, agg);
    gine_mlp_kernel<<<gemm_blocks, 128, 0, stream>>>(
        cur, agg, wmlp_p + (2 + 2 * l) * HID * HID, b1 + (size_t)l * HID,
        wmlp_p + (3 + 2 * l) * HID * HID, b2 + (size_t)l * HID, nxt, 0);
    fill_zero_kernel<<<1, 128, 0, stream>>>(stats, 128);
    bn_stats_kernel<<<stat_blocks, 256, 0, stream>>>(nxt, stats);
    bn_apply_kernel<<<zb, 256, 0, stream>>>(nxt, stats, gamma, beta, 1);
    float* t = cur; cur = nxt; nxt = t;
  }

  // ---- pool + head
  pool_kernel<<<NGRAPH, 64, 0, stream>>>(cur, pooled);
  head_kernel<<<NGRAPH, 128, 0, stream>>>(pooled, Wl1, bl1, Wl2, bl2, out);
}
